// IP_FuAd_AttnProcessor2_0_exp_9904194585116
// MI455X (gfx1250) — matrix operand order repaired and checked by reading, not compiled, not run on hardware
//
#include <hip/hip_runtime.h>
#include <hip/hip_bf16.h>

// ---------------------------------------------------------------------------
// IP-adapter cross-attention for MI455X (gfx1250, wave32, WMMA).
// Pipeline: proj_small (K/V/ipK/ipV) -> gemm<0> (Q) -> attn (fused dual
// attention + prob blend) -> gemm<1> (O proj + bias).
// All matmul work on v_wmma_f32_16x16x32_f16, fp32 accumulate, f16 staging.
// ---------------------------------------------------------------------------

typedef _Float16 v16h __attribute__((ext_vector_type(16)));
typedef float    v8f  __attribute__((ext_vector_type(8)));

union H16 { v16h v; uint4 u[2]; };

#define WMMA(a, b, c) __builtin_amdgcn_wmma_f32_16x16x32_f16( \
    false, (a), false, (b), (short)0, (c), false, false)

// A-fragment (16x32 f16, M x K): lane m=l&15; K spans hib*8..+7 and 16+hib*8..+7
static __device__ __forceinline__ v16h frag_a(const _Float16* base, int lane,
                                              int ld, int kOff) {
  const int m = lane & 15, hib = lane >> 4;
  const _Float16* p = base + (size_t)m * ld + kOff + hib * 8;
  H16 x;
  x.u[0] = *(const uint4*)(p);
  x.u[1] = *(const uint4*)(p + 16);
  return x.v;
}

// B-fragment (32x16 f16, K x N) from an N-major (transposed) array:
// lane n=l&15; K spans hib*8..+7 and 16+hib*8..+7
static __device__ __forceinline__ v16h frag_b(const _Float16* base, int lane,
                                              int ld, int kOff) {
  const int n = lane & 15, hib = lane >> 4;
  const _Float16* p = base + (size_t)n * ld + kOff + hib * 8;
  H16 x;
  x.u[0] = *(const uint4*)(p);
  x.u[1] = *(const uint4*)(p + 16);
  return x.v;
}

static __device__ __forceinline__ void coop_copy(_Float16* dst,
                                                 const _Float16* src,
                                                 int nHalves, int tid,
                                                 int nThr) {
  uint4* d = (uint4*)dst;
  const uint4* s = (const uint4*)src;
  const int n = nHalves >> 3;
  for (int i = tid; i < n; i += nThr) d[i] = s[i];
}

// ---------------------------------------------------------------------------
// Kernel 1: small projections  k = text@Wk^T, v = text@Wv^T (transposed out),
// ipk = ip@Wk_ip^T, ipv = ip@Wv_ip^T (transposed out).  One wave = one 16x16
// output tile, K loop over 2048 in chunks of 32 (64 WMMAs).
// ---------------------------------------------------------------------------
__global__ __launch_bounds__(32) void proj_small(
    const float* __restrict__ enc, const float* __restrict__ Wk,
    const float* __restrict__ Wv, const float* __restrict__ Wkip,
    const float* __restrict__ Wvip, _Float16* __restrict__ kws,
    _Float16* __restrict__ vtws, _Float16* __restrict__ ipkws,
    _Float16* __restrict__ ipvtws) {
  const int nTile = blockIdx.x;                 // 0..79 -> output col tile
  const int tTile = blockIdx.y;                 // 0..4 (text) / 0..1 (ip)
  const int proj = blockIdx.z >> 2;             // 0=k 1=v 2=ipk 3=ipv
  const int b = blockIdx.z & 3;
  if (proj >= 2 && tTile >= 2) return;

  const int lane = threadIdx.x;
  const int ln = lane & 15, hib = lane >> 4;
  const float* W = (proj == 0) ? Wk : (proj == 1) ? Wv : (proj == 2) ? Wkip : Wvip;

  // A row for this lane (rows beyond valid token count are garbage; masked in
  // softmax later).  Clamp to stay in-bounds of the 97-row encoder tensor.
  int erow = (proj < 2) ? (tTile * 16 + ln) : (77 + tTile * 16 + ln);
  if (erow > 96) erow = 96;
  const float* arow = enc + ((size_t)b * 97 + erow) * 2048;
  const int n = nTile * 16 + ln;                // output column (0..1279)
  const float* brow = W + (size_t)n * 2048;

  v8f acc = {};
  for (int kc = 0; kc < 64; ++kc) {
    const int k0 = kc * 32;
    const float* p1 = arow + k0 + hib * 8;
    v16h a;
#pragma unroll
    for (int e = 0; e < 8; ++e) a[e] = (_Float16)p1[e];
#pragma unroll
    for (int e = 0; e < 8; ++e) a[8 + e] = (_Float16)p1[16 + e];
    const float* p2 = brow + k0 + hib * 8;
    v16h bb;
#pragma unroll
    for (int e = 0; e < 8; ++e) bb[e] = (_Float16)p2[e];
#pragma unroll
    for (int e = 0; e < 8; ++e) bb[8 + e] = (_Float16)p2[16 + e];
    acc = WMMA(a, bb, acc);
    if (kc + 1 < 64) {
      __builtin_prefetch(p1 + 32, 0, 0);
      __builtin_prefetch(p2 + 32, 0, 0);
    }
  }

  const int hh = n >> 6, d = n & 63;
  const size_t bh = (size_t)(b * 20 + hh);
#pragma unroll
  for (int r = 0; r < 8; ++r) {
    const int t = tTile * 16 + r + 8 * hib;     // row of output tile
    const _Float16 val = (_Float16)acc[r];
    if (proj == 0)
      kws[(bh * 80 + t) * 64 + d] = val;        // K: [b][h][80][64]
    else if (proj == 1)
      vtws[(bh * 64 + d) * 96 + t] = val;       // V^T: [b][h][64][96]
    else if (proj == 2)
      ipkws[(bh * 32 + t) * 64 + d] = val;      // ipK: [b][h][32][64]
    else
      ipvtws[(bh * 64 + d) * 32 + t] = val;     // ipV^T: [b][h][64][32]
  }
}

// ---------------------------------------------------------------------------
// Kernel 2/4: big GEMM  Y = X @ W^T (K = N = 1280).
// MODE 0: X fp32 (hidden_states), store f16 in [b][h][s][d] head layout (Q).
// MODE 1: X f16 (attention out), store fp32 + bias to d_out (O projection).
// 256 thr = 8 waves; block tile 128x128x32; wave tile 64x32 (8 WMMA/k-step).
// ---------------------------------------------------------------------------
template <int MODE>
__global__ __launch_bounds__(256) void gemm1280(
    const float* __restrict__ Af, const _Float16* __restrict__ Ah,
    const float* __restrict__ W, const float* __restrict__ bias,
    _Float16* __restrict__ outH, float* __restrict__ outF) {
  __shared__ __align__(16) _Float16 aT[128 * 32];
  __shared__ __align__(16) _Float16 bT[128 * 32];

  const int tid = threadIdx.x;
  const int lane = tid & 31, wave = tid >> 5;
  const int wm = wave & 1, wn = wave >> 1;      // 2 (M) x 4 (N) wave grid
  const int ln = lane & 15, hib = lane >> 4;
  const int nBase = blockIdx.x * 128, mBase = blockIdx.y * 128;

  v8f zero = {};
  v8f acc[4][2];
#pragma unroll
  for (int i = 0; i < 4; ++i)
#pragma unroll
    for (int j = 0; j < 2; ++j) acc[i][j] = zero;

  const int lr = tid >> 1;                      // 0..127: staged row
  const int kh = (tid & 1) * 16;                // k half within 32-chunk

  for (int kc = 0; kc < 40; ++kc) {
    const int k0 = kc * 32;
    // stage A tile (convert fp32 -> f16 in MODE 0)
    if (MODE == 0) {
      const float* src = Af + (size_t)(mBase + lr) * 1280 + k0 + kh;
      v16h t;
#pragma unroll
      for (int e = 0; e < 16; ++e) t[e] = (_Float16)src[e];
      H16 x; x.v = t;
      uint4* dst = (uint4*)(aT + lr * 32 + kh);
      dst[0] = x.u[0]; dst[1] = x.u[1];
      if (kc + 1 < 40) __builtin_prefetch(src + 32, 0, 0);
    } else {
      const _Float16* src = Ah + (size_t)(mBase + lr) * 1280 + k0 + kh;
      uint4* dst = (uint4*)(aT + lr * 32 + kh);
      dst[0] = ((const uint4*)src)[0];
      dst[1] = ((const uint4*)src)[1];
      if (kc + 1 < 40) __builtin_prefetch(src + 32, 0, 0);
    }
    // stage B tile (W row n = output col, fp32 -> f16)
    {
      const float* src = W + (size_t)(nBase + lr) * 1280 + k0 + kh;
      v16h t;
#pragma unroll
      for (int e = 0; e < 16; ++e) t[e] = (_Float16)src[e];
      H16 x; x.v = t;
      uint4* dst = (uint4*)(bT + lr * 32 + kh);
      dst[0] = x.u[0]; dst[1] = x.u[1];
      if (kc + 1 < 40) __builtin_prefetch(src + 32, 0, 0);
    }
    __syncthreads();

    v16h bfr[2];
#pragma unroll
    for (int j = 0; j < 2; ++j)
      bfr[j] = frag_b(bT + (wn * 32 + j * 16) * 32, lane, 32, 0);
#pragma unroll
    for (int i = 0; i < 4; ++i) {
      v16h af = frag_a(aT + (wm * 64 + i * 16) * 32, lane, 32, 0);
#pragma unroll
      for (int j = 0; j < 2; ++j) acc[i][j] = WMMA(af, bfr[j], acc[i][j]);
    }
    __syncthreads();
  }

#pragma unroll
  for (int i = 0; i < 4; ++i) {
#pragma unroll
    for (int j = 0; j < 2; ++j) {
      const int col = nBase + wn * 32 + j * 16 + ln;
#pragma unroll
      for (int r = 0; r < 8; ++r) {
        const int row = mBase + wm * 64 + i * 16 + r + 8 * hib;
        if (MODE == 0) {
          const int bb = row >> 12, s = row & 4095;
          const int hh = col >> 6, d = col & 63;
          outH[(((size_t)(bb * 20 + hh)) * 4096 + s) * 64 + d] =
              (_Float16)acc[i][j][r];
        } else {
          outF[(size_t)row * 1280 + col] = acc[i][j][r] + bias[col];
        }
      }
    }
  }
}

// ---------------------------------------------------------------------------
// Kernel 3: fused dual attention with prob blending.
// Block = (head, batch-pair, 64 Q rows); 4 waves x 16-row Q tiles.
// Phase 0: batch 2p (probs kept in registers, raw-only output).
// Phase 1: batch 2p+1 (probs blended 0.5*(p0+p1), output raw + ip_hs).
// ---------------------------------------------------------------------------
__global__ __launch_bounds__(128) void attn_kernel(
    const _Float16* __restrict__ qws, const _Float16* __restrict__ kws,
    const _Float16* __restrict__ vtws, const _Float16* __restrict__ ipkws,
    const _Float16* __restrict__ ipvtws, _Float16* __restrict__ hsf16) {
  __shared__ __align__(16) _Float16 kbuf[80 * 64];    // K rows t, cols d
  __shared__ __align__(16) _Float16 vtbuf[64 * 96];   // V^T rows d, cols t
  __shared__ __align__(16) _Float16 ipkbuf[32 * 64];
  __shared__ __align__(16) _Float16 ipvtbuf[64 * 32];
  __shared__ __align__(16) _Float16 pbuf[4 * 16 * 96];  // per-wave probs
  __shared__ __align__(16) _Float16 ippbuf[4 * 16 * 32];

  const int tid = threadIdx.x;
  const int lane = tid & 31, wave = tid >> 5;
  const int ln = lane & 15, hib = lane >> 4;
  const int h = blockIdx.y;
  const int pair = blockIdx.z;
  const int qt0 = blockIdx.x * 64 + wave * 16;

  _Float16* pb = pbuf + wave * 16 * 96;
  _Float16* ipb = ippbuf + wave * 16 * 32;

  float p0t[5][8];   // phase-0 text probs (for blend)
  float p0i[2][8];   // phase-0 ip probs

  for (int phase = 0; phase < 2; ++phase) {
    const int b = pair * 2 + phase;
    const size_t bh = (size_t)(b * 20 + h);

    coop_copy(kbuf,    kws    + bh * 80 * 64, 80 * 64, tid, 128);
    coop_copy(vtbuf,   vtws   + bh * 64 * 96, 64 * 96, tid, 128);
    coop_copy(ipkbuf,  ipkws  + bh * 32 * 64, 32 * 64, tid, 128);
    coop_copy(ipvtbuf, ipvtws + bh * 64 * 32, 64 * 32, tid, 128);
    __syncthreads();

    // Q fragments (two 32-wide K chunks over head_dim 64)
    const _Float16* qbase = qws + (bh * 4096 + qt0) * 64;
    v16h qa0 = frag_a(qbase, lane, 64, 0);
    v16h qa1 = frag_a(qbase, lane, 64, 32);

    // ---- text scores: 16 x 80 (77 valid) -----------------------------------
    float st[5][8];
#pragma unroll
    for (int t = 0; t < 5; ++t) {
      v8f acc = {};
      acc = WMMA(qa0, frag_b(kbuf + t * 16 * 64, lane, 64, 0), acc);
      acc = WMMA(qa1, frag_b(kbuf + t * 16 * 64, lane, 64, 32), acc);
      const bool valid = (t * 16 + ln) < 77;
#pragma unroll
      for (int r = 0; r < 8; ++r)
        st[t][r] = valid ? acc[r] * 0.125f : -3.0e38f;
    }
    // row softmax: reduce across 5 tiles + 16 lanes (half-wave shuffles)
    float mx[8], sm[8];
#pragma unroll
    for (int r = 0; r < 8; ++r) {
      float m = st[0][r];
#pragma unroll
      for (int t = 1; t < 5; ++t) m = fmaxf(m, st[t][r]);
      mx[r] = m;
    }
#pragma unroll
    for (int off = 8; off >= 1; off >>= 1)
#pragma unroll
      for (int r = 0; r < 8; ++r)
        mx[r] = fmaxf(mx[r], __shfl_xor(mx[r], off, 32));
#pragma unroll
    for (int r = 0; r < 8; ++r) sm[r] = 0.0f;
#pragma unroll
    for (int t = 0; t < 5; ++t)
#pragma unroll
      for (int r = 0; r < 8; ++r) {
        st[t][r] = __expf(st[t][r] - mx[r]);
        sm[r] += st[t][r];
      }
#pragma unroll
    for (int off = 8; off >= 1; off >>= 1)
#pragma unroll
      for (int r = 0; r < 8; ++r) sm[r] += __shfl_xor(sm[r], off, 32);
#pragma unroll
    for (int r = 0; r < 8; ++r) sm[r] = 1.0f / sm[r];
#pragma unroll
    for (int t = 0; t < 5; ++t)
#pragma unroll
      for (int r = 0; r < 8; ++r) st[t][r] *= sm[r];

    // ---- ip scores: 16 x 32 (20 valid) -------------------------------------
    float si[2][8];
#pragma unroll
    for (int t = 0; t < 2; ++t) {
      v8f acc = {};
      acc = WMMA(qa0, frag_b(ipkbuf + t * 16 * 64, lane, 64, 0), acc);
      acc = WMMA(qa1, frag_b(ipkbuf + t * 16 * 64, lane, 64, 32), acc);
      const bool valid = (t * 16 + ln) < 20;
#pragma unroll
      for (int r = 0; r < 8; ++r)
        si[t][r] = valid ? acc[r] * 0.125f : -3.0e38f;
    }
#pragma unroll
    for (int r = 0; r < 8; ++r) mx[r] = fmaxf(si[0][r], si[1][r]);
#pragma unroll
    for (int off = 8; off >= 1; off >>= 1)
#pragma unroll
      for (int r = 0; r < 8; ++r)
        mx[r] = fmaxf(mx[r], __shfl_xor(mx[r], off, 32));
#pragma unroll
    for (int r = 0; r < 8; ++r) sm[r] = 0.0f;
#pragma unroll
    for (int t = 0; t < 2; ++t)
#pragma unroll
      for (int r = 0; r < 8; ++r) {
        si[t][r] = __expf(si[t][r] - mx[r]);
        sm[r] += si[t][r];
      }
#pragma unroll
    for (int off = 8; off >= 1; off >>= 1)
#pragma unroll
      for (int r = 0; r < 8; ++r) sm[r] += __shfl_xor(sm[r], off, 32);
#pragma unroll
    for (int r = 0; r < 8; ++r) sm[r] = 1.0f / sm[r];
#pragma unroll
    for (int t = 0; t < 2; ++t)
#pragma unroll
      for (int r = 0; r < 8; ++r) si[t][r] *= sm[r];

    // ---- blend (D-layout identical across batches: pure register op) -------
    if (phase == 0) {
#pragma unroll
      for (int t = 0; t < 5; ++t)
#pragma unroll
        for (int r = 0; r < 8; ++r) p0t[t][r] = st[t][r];
#pragma unroll
      for (int t = 0; t < 2; ++t)
#pragma unroll
        for (int r = 0; r < 8; ++r) p0i[t][r] = si[t][r];
    } else {
#pragma unroll
      for (int t = 0; t < 5; ++t)
#pragma unroll
        for (int r = 0; r < 8; ++r) st[t][r] = 0.5f * (st[t][r] + p0t[t][r]);
#pragma unroll
      for (int t = 0; t < 2; ++t)
#pragma unroll
        for (int r = 0; r < 8; ++r) si[t][r] = 0.5f * (si[t][r] + p0i[t][r]);
    }

    // ---- probs D-layout -> LDS (A-layout source for P@V) -------------------
#pragma unroll
    for (int t = 0; t < 5; ++t)
#pragma unroll
      for (int r = 0; r < 8; ++r)
        pb[(r + 8 * hib) * 96 + t * 16 + ln] = (_Float16)st[t][r];
#pragma unroll
    for (int r = 0; r < 8; ++r)                    // zero pad cols 80..95
      pb[(r + 8 * hib) * 96 + 80 + ln] = (_Float16)0.0f;
#pragma unroll
    for (int t = 0; t < 2; ++t)
#pragma unroll
      for (int r = 0; r < 8; ++r)
        ipb[(r + 8 * hib) * 32 + t * 16 + ln] = (_Float16)si[t][r];
    __syncthreads();

    // ---- P @ V (text) + optional ip P @ ipV --------------------------------
    v16h pa[3];
#pragma unroll
    for (int c = 0; c < 3; ++c) pa[c] = frag_a(pb, lane, 96, c * 32);

    v8f o[4];
#pragma unroll
    for (int j = 0; j < 4; ++j) o[j] = (v8f){};
#pragma unroll
    for (int j = 0; j < 4; ++j)
#pragma unroll
      for (int c = 0; c < 3; ++c)
        o[j] = WMMA(pa[c], frag_b(vtbuf + j * 16 * 96, lane, 96, c * 32), o[j]);

    if (phase == 1) {  // odd batches add style-scaled ip attention (scale 1.0)
      v16h ipa = frag_a(ipb, lane, 32, 0);
#pragma unroll
      for (int j = 0; j < 4; ++j)
        o[j] = WMMA(ipa, frag_b(ipvtbuf + j * 16 * 32, lane, 32, 0), o[j]);
    }

    // ---- store attention output (row-major [b*4096+s][h*64+d], f16) --------
#pragma unroll
    for (int j = 0; j < 4; ++j) {
      const int d = j * 16 + ln;
#pragma unroll
      for (int r = 0; r < 8; ++r) {
        const int s = qt0 + r + 8 * hib;
        hsf16[((size_t)(b * 4096 + s)) * 1280 + h * 64 + d] =
            (_Float16)o[j][r];
      }
    }
    __syncthreads();  // protect shared tiles before next phase reload
  }
}

// ---------------------------------------------------------------------------
extern "C" void kernel_launch(void* const* d_in, const int* in_sizes, int n_in,
                              void* d_out, int out_size, void* d_ws,
                              size_t ws_size, hipStream_t stream) {
  const float* hidden = (const float*)d_in[0];   // (4,4096,1280)
  const float* enc    = (const float*)d_in[1];   // (4,97,2048)
  const float* Wq     = (const float*)d_in[2];   // (1280,1280)
  const float* Wk     = (const float*)d_in[3];   // (1280,2048)
  const float* Wv     = (const float*)d_in[4];   // (1280,2048)
  const float* Wkip   = (const float*)d_in[5];   // (1280,2048)
  const float* Wvip   = (const float*)d_in[6];   // (1280,2048)
  const float* Wo     = (const float*)d_in[7];   // (1280,1280)
  const float* bo     = (const float*)d_in[8];   // (1280,)
  float* out = (float*)d_out;

  _Float16* ws = (_Float16*)d_ws;
  size_t off = 0;
  _Float16* qws    = ws + off; off += (size_t)4 * 20 * 4096 * 64;  // Q f16
  _Float16* kws    = ws + off; off += (size_t)4 * 20 * 80 * 64;    // K (pad 80)
  _Float16* vtws   = ws + off; off += (size_t)4 * 20 * 64 * 96;    // V^T (pad 96)
  _Float16* ipkws  = ws + off; off += (size_t)4 * 20 * 32 * 64;    // ipK (pad 32)
  _Float16* ipvtws = ws + off; off += (size_t)4 * 20 * 64 * 32;    // ipV^T
  _Float16* hsf16  = ws + off;                                     // attn out

  proj_small<<<dim3(80, 5, 16), dim3(32), 0, stream>>>(
      enc, Wk, Wv, Wkip, Wvip, kws, vtws, ipkws, ipvtws);

  gemm1280<0><<<dim3(10, 128), dim3(256), 0, stream>>>(
      hidden, nullptr, Wq, nullptr, qws, nullptr);

  attn_kernel<<<dim3(64, 20, 2), dim3(128), 0, stream>>>(
      qws, kws, vtws, ipkws, ipvtws, hsf16);

  gemm1280<1><<<dim3(10, 128), dim3(256), 0, stream>>>(
      nullptr, hsf16, Wo, bo, nullptr, out);
}
